// ramsey_NN_61357902790656
// MI455X (gfx1250) — matrix operand order repaired and checked by reading, not compiled, not run on hardware
//
#include <hip/hip_runtime.h>
#include <hip/hip_bf16.h>
#include <math.h>

// ---------------- problem constants ----------------
#define NODES   2048
#define F_DIM   64
#define H_DIM   128
#define E_EDGES 2096128LL      // 2048*2047/2
#define N_TILES 131008         // E/16 (exact)
#define EB_BLOCKS 512
#define EB_WPB    4            // waves per block (128 threads)
#define EB_TPW    64           // tiles per wave: ceil(131008/2048)

typedef __attribute__((ext_vector_type(16))) __bf16         v16bf;
typedef __attribute__((ext_vector_type(8)))  float          v8f;
typedef __attribute__((ext_vector_type(8)))  unsigned short u16x8;

union Frag { v16bf v; u16x8 h[2]; };

__device__ __forceinline__ unsigned short f2bf(float f) {
  unsigned u = __float_as_uint(f);
  u += 0x7FFFu + ((u >> 16) & 1u);   // round-to-nearest-even
  return (unsigned short)(u >> 16);
}

// invert triangular index: edge e -> (i,j), i<j, np.triu_indices order
__device__ __forceinline__ void edge_ij(long long e, int& oi, int& oj) {
  const double dN = 4095.0;                       // 2N-1
  double disc = dN * dN - 8.0 * (double)e;
  if (disc < 0.0) disc = 0.0;
  int i = (int)((dN - sqrt(disc)) * 0.5);
  if (i < 0) i = 0;
  if (i > 2046) i = 2046;
  while (i < 2046 && (long long)(i + 1) * (4095 - (i + 1)) / 2 <= e) i++;
  while (i > 0 && (long long)i * (4095 - i) / 2 > e) i--;
  long long S = (long long)i * (4095 - i) / 2;
  oi = i;
  oj = (int)(e - S) + i + 1;
}

// ---------------- node stage ----------------
// a1 = leaky_relu(nf @ W1 + b1)   [2048 x 128]
__global__ __launch_bounds__(128) void k_lin1(const float* __restrict__ nf,
                                              const float* __restrict__ W,
                                              const float* __restrict__ b,
                                              float* __restrict__ out) {
  int h = threadIdx.x, i = blockIdx.x;
  float acc = b[h];
  for (int f = 0; f < F_DIM; ++f)
    acc = fmaf(nf[i * F_DIM + f], W[f * H_DIM + h], acc);
  out[i * H_DIM + h] = acc > 0.f ? acc : 0.01f * acc;
}

// a2 = leaky_relu(a1 @ W2p + b2p) [2048 x 128]
__global__ __launch_bounds__(128) void k_lin2(const float* __restrict__ a1,
                                              const float* __restrict__ W,
                                              const float* __restrict__ b,
                                              float* __restrict__ out) {
  int h = threadIdx.x, i = blockIdx.x;
  float acc = b[h];
  for (int k = 0; k < H_DIM; ++k)
    acc = fmaf(a1[i * H_DIM + k], W[k * H_DIM + h], acc);
  out[i * H_DIM + h] = acc > 0.f ? acc : 0.01f * acc;
}

// h3 = a2 @ W3p + b3p + nf        [2048 x 64]
__global__ __launch_bounds__(64) void k_lin3(const float* __restrict__ a2,
                                             const float* __restrict__ W,
                                             const float* __restrict__ b,
                                             const float* __restrict__ nf,
                                             float* __restrict__ out) {
  int f = threadIdx.x, i = blockIdx.x;
  float acc = b[f];
  for (int k = 0; k < H_DIM; ++k)
    acc = fmaf(a2[i * H_DIM + k], W[k * F_DIM + f], acc);
  out[i * F_DIM + f] = acc + nf[i * F_DIM + f];
}

// per-column (over 2048 rows) mean/var -> scale s = g*rsqrt(v+eps), shift sh = be - s*m
__global__ __launch_bounds__(256) void k_colstats(const float* __restrict__ a,
                                                  const float* __restrict__ g,
                                                  const float* __restrict__ be,
                                                  float* __restrict__ s,
                                                  float* __restrict__ sh) {
  __shared__ float ls[256], lq[256];
  int col = blockIdx.x, tid = threadIdx.x;
  float sm = 0.f, sq = 0.f;
  for (int r = tid; r < NODES; r += 256) {
    float v = a[r * H_DIM + col];
    sm += v; sq += v * v;
  }
  ls[tid] = sm; lq[tid] = sq;
  __syncthreads();
  for (int off = 128; off; off >>= 1) {
    if (tid < off) { ls[tid] += ls[tid + off]; lq[tid] += lq[tid + off]; }
    __syncthreads();
  }
  if (tid == 0) {
    float m = ls[0] * (1.0f / NODES);
    float var = lq[0] * (1.0f / NODES) - m * m;
    float sc = g[col] * rsqrtf(var + 1e-5f);
    s[col] = sc;
    sh[col] = be[col] - sc * m;
  }
}

// fold BN into next layer: Wp[k][h]=s[k]*W[k][h]; bp[h]=b[h]+sum_k sh[k]*W[k][h]
__global__ void k_fold(const float* __restrict__ W, const float* __restrict__ b,
                       const float* __restrict__ s, const float* __restrict__ sh,
                       float* __restrict__ Wp, float* __restrict__ bp,
                       int K, int Hout) {
  int h = threadIdx.x;
  if (h >= Hout) return;
  float acc = b[h];
  for (int k = 0; k < K; ++k) {
    float w = W[k * Hout + h];
    Wp[k * Hout + h] = s[k] * w;
    acc += sh[k] * w;
  }
  bp[h] = acc;
}

// ---------------- edge-stage prep ----------------
// pack W5 (64x128 f32) into bf16 B-fragment layout:
// pack[((t*2+kf)*32 + lane)*16 + e] = bf16(W5[(32*kf + 16*(lane/16) + e)*128 + 16*t + lane%16])
__global__ __launch_bounds__(256) void k_pack5(const float* __restrict__ W5,
                                               unsigned short* __restrict__ pack) {
  int g = blockIdx.x * 256 + threadIdx.x;           // 0..8191
  int e = g & 15, ln = (g >> 4) & 31, kf = (g >> 9) & 1, t = g >> 10;
  int K = 32 * kf + 16 * (ln >> 4) + e;
  int n = 16 * t + (ln & 15);
  pack[g] = f2bf(W5[K * H_DIM + n]);
}

__global__ void k_zero(float* __restrict__ p) {   // zero 256 floats (zsum|zsq)
  p[threadIdx.x] = 0.f;
}

// fold edge BN into W6:  W6p[n][c]=s5*W6, b6p[c]=b6+sum_n sh5*W6
__global__ __launch_bounds__(128) void k_fold6(const float* __restrict__ zsum,
                                               const float* __restrict__ zsq,
                                               const float* __restrict__ g5,
                                               const float* __restrict__ be5,
                                               const float* __restrict__ W6,
                                               const float* __restrict__ b6,
                                               float* __restrict__ W6p,
                                               float* __restrict__ b6p) {
  __shared__ float p0[128], p1[128];
  int n = threadIdx.x;
  const float invE = 1.0f / (float)E_EDGES;
  float m = zsum[n] * invE;
  float var = zsq[n] * invE - m * m;
  float sc = g5[n] * rsqrtf(var + 1e-5f);
  float sh = be5[n] - sc * m;
  float w0 = W6[n * 2], w1 = W6[n * 2 + 1];
  W6p[n * 2] = sc * w0;
  W6p[n * 2 + 1] = sc * w1;
  p0[n] = sh * w0; p1[n] = sh * w1;
  __syncthreads();
  for (int off = 64; off; off >>= 1) {
    if (n < off) { p0[n] += p0[n + off]; p1[n] += p1[n + off]; }
    __syncthreads();
  }
  if (n == 0) { b6p[0] = b6[0] + p0[0]; b6p[1] = b6[1] + p1[0]; }
}

// ---------------- edge pass 1: per-channel sum/sumsq of z = lrelu((hi*hj)@W5 + b5) ----------------
__global__ __launch_bounds__(128) void k_edge_pass1(const float* __restrict__ h3,
                                                    const unsigned short* __restrict__ w5pack,
                                                    const float* __restrict__ b5,
                                                    float* __restrict__ zsum,
                                                    float* __restrict__ zsq) {
  __shared__ unsigned short eW5[8 * 2 * 32 * 16];        // 16 KB packed B
  __shared__ unsigned short eTile[EB_WPB][16 * 64];      // per-wave 16x64 bf16 e-tile
  __shared__ float ssum[H_DIM], ssq[H_DIM];

  int tid = threadIdx.x, lane = tid & 31, wv = tid >> 5;
  for (int k = tid; k < 4096; k += 128)
    ((unsigned int*)eW5)[k] = ((const unsigned int*)w5pack)[k];
  if (tid < H_DIM) { ssum[tid] = 0.f; ssq[tid] = 0.f; }
  __syncthreads();

  int col = lane & 15, halfId = lane >> 4;
  float b5v[8];
#pragma unroll
  for (int t = 0; t < 8; ++t) b5v[t] = b5[t * 16 + col];

  float sumT[8], sqT[8];
#pragma unroll
  for (int t = 0; t < 8; ++t) { sumT[t] = 0.f; sqT[t] = 0.f; }

  int w = blockIdx.x * EB_WPB + wv;
  unsigned short* myE = eTile[wv];

  for (int it = 0; it < EB_TPW; ++it) {
    int tIdx = it * (EB_BLOCKS * EB_WPB) + w;
    bool valid = (tIdx < N_TILES);
    long long e0 = (long long)(valid ? tIdx : 0) * 16;
    int ei, ej;
    edge_ij(e0 + col, ei, ej);

    __syncthreads();  // protect previous iteration's fragment reads
    {
      const float4* hi4 = (const float4*)(h3 + ei * F_DIM + halfId * 32);
      const float4* hj4 = (const float4*)(h3 + ej * F_DIM + halfId * 32);
      int base = col * 64 + halfId * 32;
#pragma unroll
      for (int q = 0; q < 8; ++q) {
        float4 a = hi4[q], b = hj4[q];
        myE[base + 4 * q + 0] = f2bf(a.x * b.x);
        myE[base + 4 * q + 1] = f2bf(a.y * b.y);
        myE[base + 4 * q + 2] = f2bf(a.z * b.z);
        myE[base + 4 * q + 3] = f2bf(a.w * b.w);
      }
    }
    __syncthreads();

    // A fragments in contiguous k order (k = 16*halfId + s), matching packed B
    int kb = halfId * 16;
    Frag alo, ahi;
    alo.h[0] = *(u16x8*)&myE[col * 64 + kb];
    alo.h[1] = *(u16x8*)&myE[col * 64 + 8 + kb];
    ahi.h[0] = *(u16x8*)&myE[col * 64 + 32 + kb];
    ahi.h[1] = *(u16x8*)&myE[col * 64 + 40 + kb];

#pragma unroll
    for (int t = 0; t < 8; ++t) {
      Frag b0, b1;
      b0.h[0] = *(u16x8*)&eW5[((t * 2 + 0) * 32 + lane) * 16];
      b0.h[1] = *(u16x8*)&eW5[((t * 2 + 0) * 32 + lane) * 16 + 8];
      b1.h[0] = *(u16x8*)&eW5[((t * 2 + 1) * 32 + lane) * 16];
      b1.h[1] = *(u16x8*)&eW5[((t * 2 + 1) * 32 + lane) * 16 + 8];
      v8f c;
#pragma unroll
      for (int r = 0; r < 8; ++r) c[r] = b5v[t];
      c = __builtin_amdgcn_wmma_f32_16x16x32_bf16(false, alo.v, false, b0.v,
                                                  (short)0, c, false, false);
      c = __builtin_amdgcn_wmma_f32_16x16x32_bf16(false, ahi.v, false, b1.v,
                                                  (short)0, c, false, false);
      if (valid) {   // wave-uniform branch; EXEC stays full around WMMA
#pragma unroll
        for (int r = 0; r < 8; ++r) {
          float z = c[r];
          z = z > 0.f ? z : 0.01f * z;
          sumT[t] += z;
          sqT[t] += z * z;
        }
      }
    }
  }

  __syncthreads();
#pragma unroll
  for (int t = 0; t < 8; ++t) {
    atomicAdd(&ssum[t * 16 + col], sumT[t]);
    atomicAdd(&ssq[t * 16 + col], sqT[t]);
  }
  __syncthreads();
  if (tid < H_DIM) {
    atomicAdd(&zsum[tid], ssum[tid]);
    atomicAdd(&zsq[tid], ssq[tid]);
  }
}

// ---------------- edge pass 2: logits = z @ W6p + b6p, softmax, symmetric scatter ----------------
__global__ __launch_bounds__(128) void k_edge_pass2(const float* __restrict__ h3,
                                                    const unsigned short* __restrict__ w5pack,
                                                    const float* __restrict__ b5,
                                                    const float* __restrict__ W6p,
                                                    const float* __restrict__ b6p,
                                                    float* __restrict__ out) {
  __shared__ unsigned short eW5[8 * 2 * 32 * 16];
  __shared__ unsigned short eTile[EB_WPB][16 * 64];
  __shared__ float zrow[EB_WPB][16][H_DIM];            // 32 KB
  __shared__ float w6s[H_DIM * 2];
  __shared__ float b6s[2];

  int tid = threadIdx.x, lane = tid & 31, wv = tid >> 5;
  for (int k = tid; k < 4096; k += 128)
    ((unsigned int*)eW5)[k] = ((const unsigned int*)w5pack)[k];
  for (int k = tid; k < H_DIM * 2; k += 128) w6s[k] = W6p[k];
  if (tid < 2) b6s[tid] = b6p[tid];
  __syncthreads();

  int col = lane & 15, halfId = lane >> 4;
  float b5v[8];
#pragma unroll
  for (int t = 0; t < 8; ++t) b5v[t] = b5[t * 16 + col];

  int w = blockIdx.x * EB_WPB + wv;
  unsigned short* myE = eTile[wv];

  for (int it = 0; it < EB_TPW; ++it) {
    int tIdx = it * (EB_BLOCKS * EB_WPB) + w;
    bool valid = (tIdx < N_TILES);
    long long e0 = (long long)(valid ? tIdx : 0) * 16;
    int ei, ej;
    edge_ij(e0 + col, ei, ej);

    __syncthreads();
    {
      const float4* hi4 = (const float4*)(h3 + ei * F_DIM + halfId * 32);
      const float4* hj4 = (const float4*)(h3 + ej * F_DIM + halfId * 32);
      int base = col * 64 + halfId * 32;
#pragma unroll
      for (int q = 0; q < 8; ++q) {
        float4 a = hi4[q], b = hj4[q];
        myE[base + 4 * q + 0] = f2bf(a.x * b.x);
        myE[base + 4 * q + 1] = f2bf(a.y * b.y);
        myE[base + 4 * q + 2] = f2bf(a.z * b.z);
        myE[base + 4 * q + 3] = f2bf(a.w * b.w);
      }
    }
    __syncthreads();

    int kb = halfId * 16;
    Frag alo, ahi;
    alo.h[0] = *(u16x8*)&myE[col * 64 + kb];
    alo.h[1] = *(u16x8*)&myE[col * 64 + 8 + kb];
    ahi.h[0] = *(u16x8*)&myE[col * 64 + 32 + kb];
    ahi.h[1] = *(u16x8*)&myE[col * 64 + 40 + kb];

#pragma unroll
    for (int t = 0; t < 8; ++t) {
      Frag b0, b1;
      b0.h[0] = *(u16x8*)&eW5[((t * 2 + 0) * 32 + lane) * 16];
      b0.h[1] = *(u16x8*)&eW5[((t * 2 + 0) * 32 + lane) * 16 + 8];
      b1.h[0] = *(u16x8*)&eW5[((t * 2 + 1) * 32 + lane) * 16];
      b1.h[1] = *(u16x8*)&eW5[((t * 2 + 1) * 32 + lane) * 16 + 8];
      v8f c;
#pragma unroll
      for (int r = 0; r < 8; ++r) c[r] = b5v[t];
      c = __builtin_amdgcn_wmma_f32_16x16x32_bf16(false, alo.v, false, b0.v,
                                                  (short)0, c, false, false);
      c = __builtin_amdgcn_wmma_f32_16x16x32_bf16(false, ahi.v, false, b1.v,
                                                  (short)0, c, false, false);
      // scatter D into per-edge rows (D layout: M = r + 8*halfId, N = col)
#pragma unroll
      for (int r = 0; r < 8; ++r) {
        float z = c[r];
        z = z > 0.f ? z : 0.01f * z;
        zrow[wv][r + 8 * halfId][t * 16 + col] = z;
      }
    }
    __syncthreads();

    // each lane: edge m = col, channel half = halfId
    float l0 = 0.f, l1 = 0.f;
    const float* zr = &zrow[wv][col][halfId * 64];
#pragma unroll
    for (int n = 0; n < 64; ++n) {
      float z = zr[n];
      l0 = fmaf(z, w6s[(halfId * 64 + n) * 2 + 0], l0);
      l1 = fmaf(z, w6s[(halfId * 64 + n) * 2 + 1], l1);
    }
    l0 += __shfl_xor(l0, 16, 32);
    l1 += __shfl_xor(l1, 16, 32);
    if (valid && halfId == 0) {
      l0 += b6s[0];
      l1 += b6s[1];
      float mx = fmaxf(l0, l1);
      float p0 = __expf(l0 - mx), p1 = __expf(l1 - mx);
      float inv = 1.f / (p0 + p1);
      float2 pv = make_float2(p0 * inv, p1 * inv);
      *(float2*)&out[((size_t)ei * NODES + ej) * 2] = pv;
      *(float2*)&out[((size_t)ej * NODES + ei) * 2] = pv;
    }
  }
}

__global__ __launch_bounds__(256) void k_diag(float* __restrict__ out) {
  int i = blockIdx.x * 256 + threadIdx.x;
  if (i < NODES) {
    out[((size_t)i * NODES + i) * 2 + 0] = 0.f;
    out[((size_t)i * NODES + i) * 2 + 1] = 0.f;
  }
}

// ---------------- launch ----------------
extern "C" void kernel_launch(void* const* d_in, const int* in_sizes, int n_in,
                              void* d_out, int out_size, void* d_ws, size_t ws_size,
                              hipStream_t stream) {
  (void)in_sizes; (void)n_in; (void)out_size; (void)ws_size;
  const float* nf  = (const float*)d_in[1];   // node_features (d_in[0] "x" is unused by reference)
  const float* W1  = (const float*)d_in[2];
  const float* b1  = (const float*)d_in[3];
  const float* g1  = (const float*)d_in[4];
  const float* be1 = (const float*)d_in[5];
  const float* W2  = (const float*)d_in[6];
  const float* b2  = (const float*)d_in[7];
  const float* g2  = (const float*)d_in[8];
  const float* be2 = (const float*)d_in[9];
  const float* W3  = (const float*)d_in[10];
  const float* b3  = (const float*)d_in[11];
  const float* W5  = (const float*)d_in[12];
  const float* b5  = (const float*)d_in[13];
  const float* g5  = (const float*)d_in[14];
  const float* be5 = (const float*)d_in[15];
  const float* W6  = (const float*)d_in[16];
  const float* b6  = (const float*)d_in[17];
  float* out = (float*)d_out;

  // workspace layout (floats) — total ~2.75 MB
  float* ws = (float*)d_ws;
  float* a1   = ws;                 // 262144
  float* a2   = ws + 262144;        // 262144
  float* h3   = ws + 524288;        // 131072
  float* s1   = ws + 655360;        // 128
  float* sh1  = ws + 655488;        // 128
  float* s2   = ws + 655616;        // 128
  float* sh2  = ws + 655744;        // 128
  float* W2p  = ws + 655872;        // 16384
  float* b2p  = ws + 672256;        // 128
  float* W3p  = ws + 672384;        // 8192
  float* b3p  = ws + 680576;        // 64
  float* zsum = ws + 680640;        // 128 (zsq contiguous after)
  float* zsq  = ws + 680768;        // 128
  float* W6p  = ws + 680896;        // 256
  float* b6p  = ws + 681152;        // 2 (+pad)
  unsigned short* w5pk = (unsigned short*)(ws + 681160); // 8192 ushorts

  // node stage with BN folded into the following layer
  k_lin1<<<dim3(NODES), dim3(H_DIM), 0, stream>>>(nf, W1, b1, a1);
  k_colstats<<<dim3(H_DIM), dim3(256), 0, stream>>>(a1, g1, be1, s1, sh1);
  k_fold<<<dim3(1), dim3(H_DIM), 0, stream>>>(W2, b2, s1, sh1, W2p, b2p, H_DIM, H_DIM);
  k_lin2<<<dim3(NODES), dim3(H_DIM), 0, stream>>>(a1, W2p, b2p, a2);
  k_colstats<<<dim3(H_DIM), dim3(256), 0, stream>>>(a2, g2, be2, s2, sh2);
  k_fold<<<dim3(1), dim3(F_DIM), 0, stream>>>(W3, b3, s2, sh2, W3p, b3p, H_DIM, F_DIM);
  k_lin3<<<dim3(NODES), dim3(F_DIM), 0, stream>>>(a2, W3p, b3p, nf, h3);

  // edge stage
  k_pack5<<<dim3(32), dim3(256), 0, stream>>>(W5, w5pk);
  k_zero<<<dim3(1), dim3(256), 0, stream>>>(zsum);   // zeroes zsum+zsq (contiguous 256 floats)
  k_edge_pass1<<<dim3(EB_BLOCKS), dim3(128), 0, stream>>>(h3, w5pk, b5, zsum, zsq);
  k_fold6<<<dim3(1), dim3(H_DIM), 0, stream>>>(zsum, zsq, g5, be5, W6, b6, W6p, b6p);
  k_edge_pass2<<<dim3(EB_BLOCKS), dim3(128), 0, stream>>>(h3, w5pk, b5, W6p, b6p, out);
  k_diag<<<dim3(8), dim3(256), 0, stream>>>(out);
}
